// YOLOv5_31705448579261
// MI455X (gfx1250) — matrix operand order repaired and checked by reading, not compiled, not run on hardware
//
#include <hip/hip_runtime.h>

typedef __attribute__((ext_vector_type(16))) __bf16 v16bf;
typedef __attribute__((ext_vector_type(8)))  float  v8f;
typedef __attribute__((ext_vector_type(4)))  unsigned int u32x4;
typedef __attribute__((ext_vector_type(4)))  float  f32x4;

union FragB16 { v16bf v; u32x4 q[2]; };

#define DI static __device__ __forceinline__

DI unsigned short f2bf(float f) {
    unsigned int u = __float_as_uint(f);
    u += 0x7fffu + ((u >> 16) & 1u);          // round-to-nearest-even
    return (unsigned short)(u >> 16);
}
DI float bf2f(unsigned short h) { return __uint_as_float(((unsigned int)h) << 16); }
DI float hswish_f(float x) {
    float t = fminf(fmaxf(x + 3.0f, 0.0f), 6.0f);
    return x * t * (1.0f / 6.0f);
}
DI v8f vzero8() { v8f z = {0.f,0.f,0.f,0.f,0.f,0.f,0.f,0.f}; return z; }

// ---------------------------------------------------------------------------
// K1: fused cv1 (1x1 128->64, BN+hswish) and cv2p (1x1 128->64, bn_cat[64:]+leaky)
// block = 128 threads (4 waves), 64 rows/block.  LDS: 16KB A + 32KB W + 1KB sb
// ---------------------------------------------------------------------------
__global__ __launch_bounds__(128) void k_cv1_cv2p(
    const float* __restrict__ x, const float* __restrict__ w1,
    const float* __restrict__ w2, const float* __restrict__ bn1,
    const float* __restrict__ bncat,
    unsigned short* __restrict__ y0, unsigned short* __restrict__ x2)
{
    __shared__ unsigned short sA[64 * 128];
    __shared__ unsigned short sW[128 * 128];
    __shared__ float sScale[128], sBias[128];

    const int tid = threadIdx.x, lane = tid & 31, wv = tid >> 5;

    if (tid < 128) {
        float g, b, m, v;
        if (tid < 64) { g = bn1[tid];      b = bn1[64 + tid];   m = bn1[128 + tid];  v = bn1[192 + tid]; }
        else          { g = bncat[tid];    b = bncat[128 + tid]; m = bncat[256 + tid]; v = bncat[384 + tid]; }
        float s = g * rsqrtf(v + 1e-3f);
        sScale[tid] = s; sBias[tid] = b - m * s;
    }
    // prepack weights into fragment order: KST=4, NT=8
    for (int i = 0; i < 128; ++i) {
        int idx = tid + i * 128;
        int j = idx & 15, ln = (idx >> 4) & 31, rest = idx >> 9;
        int nt = rest & 7, ks = rest >> 3;
        int k = ks * 32 + ((ln >= 16) ? 8 : 0) + ((j >= 8) ? 16 : 0) + (j & 7);
        int n = nt * 16 + (ln & 15);
        float wval = (n < 64) ? w1[k * 64 + n] : w2[k * 64 + (n - 64)];
        sW[idx] = f2bf(wval);
    }
    // A tile: 64 rows x 128ch, f32 -> bf16
    const int p0 = blockIdx.x * 64;
    for (int i = 0; i < 16; ++i) {
        int idx = tid + i * 128;               // 2048 float4 chunks
        int row = idx >> 5, c4 = (idx & 31) << 2;
        f32x4 v = *(const f32x4*)(x + (long)(p0 + row) * 128 + c4);
        unsigned short* d = &sA[row * 128 + c4];
        d[0] = f2bf(v.x); d[1] = f2bf(v.y); d[2] = f2bf(v.z); d[3] = f2bf(v.w);
    }
    __syncthreads();

    v8f acc[8];
    for (int t = 0; t < 8; ++t) acc[t] = vzero8();
    const int m0 = wv * 16;
    const int khalf = (lane >= 16) ? 8 : 0;

    for (int ks = 0; ks < 4; ++ks) {
        FragB16 a;
        const unsigned short* ap = &sA[(m0 + (lane & 15)) * 128 + ks * 32 + khalf];
        a.q[0] = *(const u32x4*)ap;
        a.q[1] = *(const u32x4*)(ap + 16);
        for (int nt = 0; nt < 8; ++nt) {
            FragB16 b;
            const unsigned short* bp = &sW[((ks * 8 + nt) * 32 + lane) * 16];
            b.q[0] = *(const u32x4*)bp;
            b.q[1] = *(const u32x4*)(bp + 8);
            acc[nt] = __builtin_amdgcn_wmma_f32_16x16x32_bf16(
                false, a.v, false, b.v, (short)0, acc[nt], false, false);
        }
    }
    const int mrow = (lane >> 4) << 3;
    for (int nt = 0; nt < 8; ++nt) {
        int n = nt * 16 + (lane & 15);
        float s = sScale[n], bb = sBias[n];
        for (int r = 0; r < 8; ++r) {
            long p = p0 + m0 + mrow + r;
            float v = acc[nt][r] * s + bb;
            if (n < 64) { v = hswish_f(v);               y0[p * 64 + n]        = f2bf(v); }
            else        { v = (v >= 0.f) ? v : 0.1f * v; x2[p * 64 + (n - 64)] = f2bf(v); }
        }
    }
}

// ---------------------------------------------------------------------------
// Generic 1x1 64->64 GEMM, bf16 in/out.  act: 0 = hardswish, 1 = leaky(0.1)
// block = 256 (8 waves), 128 rows/block.  LDS: 16KB A + 8KB W + 0.5KB sb
// ---------------------------------------------------------------------------
__global__ __launch_bounds__(256) void k_gemm64(
    const unsigned short* __restrict__ in, const float* __restrict__ w,
    const float* __restrict__ bn, int bnStride, int act,
    unsigned short* __restrict__ out)
{
    __shared__ unsigned short sA[128 * 64];
    __shared__ unsigned short sW[64 * 64];
    __shared__ float sScale[64], sBias[64];

    const int tid = threadIdx.x, lane = tid & 31, wv = tid >> 5;

    if (tid < 64) {
        float g = bn[tid], b = bn[bnStride + tid];
        float m = bn[2 * bnStride + tid], v = bn[3 * bnStride + tid];
        float s = g * rsqrtf(v + 1e-3f);
        sScale[tid] = s; sBias[tid] = b - m * s;
    }
    for (int i = 0; i < 16; ++i) {             // prepack KST=2, NT=4 (4096 elems)
        int idx = tid + i * 256;
        int j = idx & 15, ln = (idx >> 4) & 31, rest = idx >> 9;
        int nt = rest & 3, ks = rest >> 2;
        int k = ks * 32 + ((ln >= 16) ? 8 : 0) + ((j >= 8) ? 16 : 0) + (j & 7);
        int n = nt * 16 + (ln & 15);
        sW[idx] = f2bf(w[k * 64 + n]);
    }
    const int p0 = blockIdx.x * 128;
    for (int i = 0; i < 4; ++i) {              // 1024 x 8-elem chunks
        int idx = tid + i * 256;
        int row = idx >> 3, c8 = (idx & 7) << 3;
        *(u32x4*)&sA[row * 64 + c8] = *(const u32x4*)(in + (long)(p0 + row) * 64 + c8);
    }
    __syncthreads();

    v8f acc[4];
    for (int t = 0; t < 4; ++t) acc[t] = vzero8();
    const int m0 = wv * 16;
    const int khalf = (lane >= 16) ? 8 : 0;

    for (int ks = 0; ks < 2; ++ks) {
        FragB16 a;
        const unsigned short* ap = &sA[(m0 + (lane & 15)) * 64 + ks * 32 + khalf];
        a.q[0] = *(const u32x4*)ap;
        a.q[1] = *(const u32x4*)(ap + 16);
        for (int nt = 0; nt < 4; ++nt) {
            FragB16 b;
            const unsigned short* bp = &sW[((ks * 4 + nt) * 32 + lane) * 16];
            b.q[0] = *(const u32x4*)bp;
            b.q[1] = *(const u32x4*)(bp + 8);
            acc[nt] = __builtin_amdgcn_wmma_f32_16x16x32_bf16(
                false, a.v, false, b.v, (short)0, acc[nt], false, false);
        }
    }
    const int mrow = (lane >> 4) << 3;
    for (int nt = 0; nt < 4; ++nt) {
        int n = nt * 16 + (lane & 15);
        float s = sScale[n], bb = sBias[n];
        for (int r = 0; r < 8; ++r) {
            long p = p0 + m0 + mrow + r;
            float v = acc[nt][r] * s + bb;
            v = (act == 0) ? hswish_f(v) : ((v >= 0.f) ? v : 0.1f * v);
            out[p * 64 + n] = f2bf(v);
        }
    }
}

// ---------------------------------------------------------------------------
// 3x3 SAME conv 64->64 as 9 shifted WMMA GEMMs + BN + hswish + residual add.
// block = 128 (4 waves), 64 output x-positions / block; grid (3,160,16).
// LDS: tile 3x68x64 bf16 (25.5KB) + 3-tap staged weights (24KB) + sb
// ---------------------------------------------------------------------------
__global__ __launch_bounds__(128) void k_conv3x3(
    const unsigned short* __restrict__ in, const float* __restrict__ w3,
    const float* __restrict__ bn, const unsigned short* __restrict__ resid,
    unsigned short* __restrict__ out)
{
    __shared__ unsigned short sT[3 * 68 * 64];
    __shared__ unsigned short sW[3 * 64 * 64];
    __shared__ float sScale[64], sBias[64];

    const int tid = threadIdx.x, lane = tid & 31, wv = tid >> 5;
    const int x0 = blockIdx.x * 64;
    const int yy = blockIdx.y;
    const int img = blockIdx.z;

    if (tid < 64) {
        float g = bn[tid], b = bn[64 + tid], m = bn[128 + tid], v = bn[192 + tid];
        float s = g * rsqrtf(v + 1e-3f);
        sScale[tid] = s; sBias[tid] = b - m * s;
    }
    // halo tile: 3 rows x 66 x-positions x 64ch (zero-padded OOB)
    for (int i = 0; i < 13; ++i) {
        int idx = tid + i * 128;
        if (idx < 3 * 66 * 8) {
            int cc = (idx & 7) << 3;
            int pos = idx >> 3;
            int dy = pos / 66, xi = pos % 66;
            int gy = yy - 1 + dy, gx = x0 - 1 + xi;
            u32x4 val = {0u, 0u, 0u, 0u};
            if ((unsigned)gy < 160u && (unsigned)gx < 160u)
                val = *(const u32x4*)(in + (((long)img * 160 + gy) * 160 + gx) * 64 + cc);
            *(u32x4*)&sT[(dy * 68 + xi) * 64 + cc] = val;
        }
    }

    v8f acc[4];
    for (int t = 0; t < 4; ++t) acc[t] = vzero8();
    const int khalf = (lane >= 16) ? 8 : 0;

    for (int dy = 0; dy < 3; ++dy) {
        __syncthreads();                       // also covers initial tile/sb writes
        for (int i = 0; i < 96; ++i) {         // stage 3 taps of this dy (12288 elems)
            int idx = tid + i * 128;
            int j = idx & 15, ln = (idx >> 4) & 31, rest = idx >> 9;
            int nt = rest & 3, ks = (rest >> 2) & 1, dx = rest >> 3;
            int k = ks * 32 + ((ln >= 16) ? 8 : 0) + ((j >= 8) ? 16 : 0) + (j & 7);
            int n = nt * 16 + (ln & 15);
            sW[idx] = f2bf(w3[(((dy * 3 + dx) * 64) + k) * 64 + n]);
        }
        __syncthreads();
        for (int dx = 0; dx < 3; ++dx) {
            for (int ks = 0; ks < 2; ++ks) {
                FragB16 a;
                int xi = wv * 16 + (lane & 15) + dx;
                const unsigned short* ap = &sT[(dy * 68 + xi) * 64 + ks * 32 + khalf];
                a.q[0] = *(const u32x4*)ap;
                a.q[1] = *(const u32x4*)(ap + 16);
                for (int nt = 0; nt < 4; ++nt) {
                    FragB16 b;
                    const unsigned short* bp = &sW[(((dx * 2 + ks) * 4 + nt) * 32 + lane) * 16];
                    b.q[0] = *(const u32x4*)bp;
                    b.q[1] = *(const u32x4*)(bp + 8);
                    acc[nt] = __builtin_amdgcn_wmma_f32_16x16x32_bf16(
                        false, a.v, false, b.v, (short)0, acc[nt], false, false);
                }
            }
        }
    }
    const int mrow = (lane >> 4) << 3;
    for (int nt = 0; nt < 4; ++nt) {
        int n = nt * 16 + (lane & 15);
        float s = sScale[n], bb = sBias[n];
        for (int r = 0; r < 8; ++r) {
            int px = x0 + wv * 16 + mrow + r;
            if (px < 160) {
                long p = ((long)img * 160 + yy) * 160 + px;
                float v = hswish_f(acc[nt][r] * s + bb) + bf2f(resid[p * 64 + n]);
                out[p * 64 + n] = f2bf(v);
            }
        }
    }
}

// ---------------------------------------------------------------------------
// Final cv4: A = [x1act | x2act] (both pre-activated bf16), K=128 -> N=128,
// BN + hswish, fp32 output.  block = 128 (4 waves), 64 rows/block.
// ---------------------------------------------------------------------------
__global__ __launch_bounds__(128) void k_final(
    const unsigned short* __restrict__ x1a, const unsigned short* __restrict__ x2a,
    const float* __restrict__ w4, const float* __restrict__ bn4,
    float* __restrict__ out)
{
    __shared__ unsigned short sA[64 * 128];
    __shared__ unsigned short sW[128 * 128];
    __shared__ float sScale[128], sBias[128];

    const int tid = threadIdx.x, lane = tid & 31, wv = tid >> 5;

    if (tid < 128) {
        float g = bn4[tid], b = bn4[128 + tid], m = bn4[256 + tid], v = bn4[384 + tid];
        float s = g * rsqrtf(v + 1e-3f);
        sScale[tid] = s; sBias[tid] = b - m * s;
    }
    for (int i = 0; i < 128; ++i) {            // prepack KST=4, NT=8 (16384 elems)
        int idx = tid + i * 128;
        int j = idx & 15, ln = (idx >> 4) & 31, rest = idx >> 9;
        int nt = rest & 7, ks = rest >> 3;
        int k = ks * 32 + ((ln >= 16) ? 8 : 0) + ((j >= 8) ? 16 : 0) + (j & 7);
        int n = nt * 16 + (ln & 15);
        sW[idx] = f2bf(w4[k * 128 + n]);
    }
    const int p0 = blockIdx.x * 64;
    for (int i = 0; i < 8; ++i) {              // 1024 x 8-elem chunks
        int idx = tid + i * 128;
        int row = idx >> 4, c8 = (idx & 15) << 3;
        u32x4 val = (c8 < 64)
            ? *(const u32x4*)(x1a + (long)(p0 + row) * 64 + c8)
            : *(const u32x4*)(x2a + (long)(p0 + row) * 64 + (c8 - 64));
        *(u32x4*)&sA[row * 128 + c8] = val;
    }
    __syncthreads();

    v8f acc[8];
    for (int t = 0; t < 8; ++t) acc[t] = vzero8();
    const int m0 = wv * 16;
    const int khalf = (lane >= 16) ? 8 : 0;

    for (int ks = 0; ks < 4; ++ks) {
        FragB16 a;
        const unsigned short* ap = &sA[(m0 + (lane & 15)) * 128 + ks * 32 + khalf];
        a.q[0] = *(const u32x4*)ap;
        a.q[1] = *(const u32x4*)(ap + 16);
        for (int nt = 0; nt < 8; ++nt) {
            FragB16 b;
            const unsigned short* bp = &sW[((ks * 8 + nt) * 32 + lane) * 16];
            b.q[0] = *(const u32x4*)bp;
            b.q[1] = *(const u32x4*)(bp + 8);
            acc[nt] = __builtin_amdgcn_wmma_f32_16x16x32_bf16(
                false, a.v, false, b.v, (short)0, acc[nt], false, false);
        }
    }
    const int mrow = (lane >> 4) << 3;
    for (int nt = 0; nt < 8; ++nt) {
        int n = nt * 16 + (lane & 15);
        float s = sScale[n], bb = sBias[n];
        for (int r = 0; r < 8; ++r) {
            long p = p0 + m0 + mrow + r;
            out[p * 128 + n] = hswish_f(acc[nt][r] * s + bb);
        }
    }
}

// ---------------------------------------------------------------------------
extern "C" void kernel_launch(void* const* d_in, const int* in_sizes, int n_in,
                              void* d_out, int out_size, void* d_ws, size_t ws_size,
                              hipStream_t stream) {
    const float* x        = (const float*)d_in[0];
    const float* cv1_w    = (const float*)d_in[1];
    const float* cv1_bn   = (const float*)d_in[2];
    const float* m0c1_w   = (const float*)d_in[3];
    const float* m0c1_bn  = (const float*)d_in[4];
    const float* m0c2_w   = (const float*)d_in[5];
    const float* m0c2_bn  = (const float*)d_in[6];
    const float* m1c1_w   = (const float*)d_in[7];
    const float* m1c1_bn  = (const float*)d_in[8];
    const float* m1c2_w   = (const float*)d_in[9];
    const float* m1c2_bn  = (const float*)d_in[10];
    const float* cv3_w    = (const float*)d_in[11];
    const float* cv2p_w   = (const float*)d_in[12];
    const float* bn_cat   = (const float*)d_in[13];
    const float* cv4_w    = (const float*)d_in[14];
    const float* cv4_bn   = (const float*)d_in[15];

    const size_t SLOT = 409600ull * 64;        // bf16 elements per slot
    unsigned short* s0 = (unsigned short*)d_ws;
    unsigned short* s1 = s0 + SLOT;
    unsigned short* s2 = s0 + 2 * SLOT;
    unsigned short* s3 = s0 + 3 * SLOT;

    // 1) y0 = hswish(bn(cv1(x)))  [s0];  x2act = leaky(bn_cat[64:](cv2p(x))) [s3]
    k_cv1_cv2p<<<6400, 128, 0, stream>>>(x, cv1_w, cv2p_w, cv1_bn, bn_cat, s0, s3);
    // 2) h0 = hswish(bn(1x1(y0)))  [s1]
    k_gemm64<<<3200, 256, 0, stream>>>(s0, m0c1_w, m0c1_bn, 64, 0, s1);
    // 3) y1 = y0 + hswish(bn(3x3(h0)))  [s2]
    dim3 g3(3, 160, 16);
    k_conv3x3<<<g3, 128, 0, stream>>>(s1, m0c2_w, m0c2_bn, s0, s2);
    // 4) h1 = hswish(bn(1x1(y1)))  [s1]
    k_gemm64<<<3200, 256, 0, stream>>>(s2, m1c1_w, m1c1_bn, 64, 0, s1);
    // 5) y2 = y1 + hswish(bn(3x3(h1)))  [s0]
    k_conv3x3<<<g3, 128, 0, stream>>>(s1, m1c2_w, m1c2_bn, s2, s0);
    // 6) x1act = leaky(bn_cat[0:64](cv3(y2)))  [s2]
    k_gemm64<<<3200, 256, 0, stream>>>(s0, cv3_w, bn_cat, 128, 1, s2);
    // 7) out = hswish(bn4(cv4([x1act|x2act])))  fp32
    k_final<<<6400, 128, 0, stream>>>(s2, s3, cv4_w, cv4_bn, (float*)d_out);
}
